// BasicBlock_19533511262377
// MI455X (gfx1250) — compile-verified, matrix operand order checked by reading, not run on hardware
//
#include <hip/hip_runtime.h>

// ---------------------------------------------------------------------------
// Types for CDNA5 WMMA (wave32): A/B = 16 x bf16 per lane, C/D = 8 x f32.
// ---------------------------------------------------------------------------
typedef __bf16 v16bf __attribute__((ext_vector_type(16)));
typedef float  v8f   __attribute__((ext_vector_type(8)));
typedef unsigned int v4u __attribute__((ext_vector_type(4)));
typedef int  v8i __attribute__((ext_vector_type(8)));
typedef int  v4i __attribute__((ext_vector_type(4)));

union FragBF {
    v16bf v;
    struct { uint4 lo, hi; } u;   // two 16B halves (8 VGPRs)
};

// ---------------------------------------------------------------------------
// CDNA5 Tensor Data Mover: one DMA per stage moves a 2D weight tile
// (rows x tile_k bf16, row stride row_stride) from global into contiguous LDS.
// D# built per CDNA5 ISA. Tracked by TENSORcnt.
// This toolchain's builtin takes 6 args (g0, g1, g2, g3, g4, cpol).
// ---------------------------------------------------------------------------
__device__ __forceinline__ void tdm_load_2d_bf16(void* lds_ptr, const void* gptr,
                                                 int tile_k, int rows,
                                                 int row_stride)
{
    unsigned long long ga = (unsigned long long)(uintptr_t)gptr;
    v4u g0;
    g0.x = 1u;                                        // count=1 (valid), user mode
    g0.y = (unsigned)(uintptr_t)lds_ptr;              // lds_addr (low 32b generic)
    g0.z = (unsigned)(ga & 0xFFFFFFFFu);              // global_addr[31:0]
    g0.w = (unsigned)((ga >> 32) & 0x01FFFFFFu)       // global_addr[56:32]
         | 0x80000000u;                               // type=2 ("image")
    v8i g1;
    g1[0] = 1 << 16;                                  // data_size=1 (2 bytes)
    g1[1] = (row_stride & 0xFFFF) << 16;              // tensor_dim0[15:0]
    g1[2] = ((row_stride >> 16) & 0xFFFF)             // tensor_dim0[31:16]
          | ((rows & 0xFFFF) << 16);                  // tensor_dim1[15:0]
    g1[3] = ((rows >> 16) & 0xFFFF)                   // tensor_dim1[31:16]
          | (tile_k << 16);                           // tile_dim0
    g1[4] = rows & 0xFFFF;                            // tile_dim1 (tile_dim2=0)
    g1[5] = row_stride;                               // tensor_dim0_stride[31:0]
    g1[6] = 0;                                        // dim0_stride[47:32]=0, dim1_stride lo=0
    g1[7] = 0;                                        // tensor_dim1_stride unused
    v4i z4 = (v4i)(0);
    v8i z8 = (v8i)(0);
    __builtin_amdgcn_tensor_load_to_lds(g0, g1, z4, z4, z8, 0);
}
__device__ __forceinline__ void wait_tensorcnt0()
{
    __builtin_amdgcn_s_wait_tensorcnt(0);
}

// ---------------------------------------------------------------------------
// HBFP8 block quantizer (block = 64 contiguous elements shares one exponent).
// One wave per block: lane handles elements i and i+32. Optionally fuses
// training-mode BN (+ReLU) before quantizing. bf16 output is exact (<=8
// significant mantissa bits in HBFP8). All index math is 32-bit.
// ---------------------------------------------------------------------------
__global__ __launch_bounds__(256)
void k_bfp_quant(const float* __restrict__ in, __bf16* __restrict__ out,
                 unsigned n,
                 const float* __restrict__ mstd,   // [2*C] (mean,istd) or null
                 const float* __restrict__ gamma,
                 const float* __restrict__ beta,
                 unsigned C, unsigned HW, int do_relu)
{
    const unsigned lane  = threadIdx.x & 31;
    const unsigned nblk  = (n + 63) >> 6;
    unsigned wid         = blockIdx.x * (blockDim.x >> 5) + (threadIdx.x >> 5);
    const unsigned wstep = gridDim.x * (blockDim.x >> 5);

    for (unsigned b = wid; b < nblk; b += wstep) {
        unsigned i0 = (b << 6) + lane;
        unsigned i1 = i0 + 32;
        float v0 = 0.f, v1 = 0.f;
        if (i0 < n) v0 = in[i0];
        if (i1 < n) v1 = in[i1];
        if (mstd) {
            unsigned c0 = (i0 / HW) % C;
            unsigned c1 = (i1 / HW) % C;
            v0 = (v0 - mstd[2*c0]) * mstd[2*c0+1] * gamma[c0] + beta[c0];
            v1 = (v1 - mstd[2*c1]) * mstd[2*c1+1] * gamma[c1] + beta[c1];
            if (do_relu) { v0 = fmaxf(v0, 0.f); v1 = fmaxf(v1, 0.f); }
        }
        float m = fmaxf(fabsf(v0), fabsf(v1));
        #pragma unroll
        for (int off = 16; off > 0; off >>= 1)
            m = fmaxf(m, __shfl_xor(m, off, 32));

        float e        = ceilf(log2f(m + 1e-23f));
        float interval = exp2f(e - 8.0f);
        float maxv     = exp2f(e) - interval;
        float rinv     = 1.0f / interval;
        float q0 = fminf(fmaxf(rintf(v0 * rinv) * interval, -maxv), maxv);
        float q1 = fminf(fmaxf(rintf(v1 * rinv) * interval, -maxv), maxv);
        if (i0 < n) out[i0] = (__bf16)q0;
        if (i1 < n) out[i1] = (__bf16)q1;
    }
}

// ---------------------------------------------------------------------------
// Implicit-GEMM conv on WMMA bf16 (wave32), compile-time filter geometry.
//   Tile BM=128, BN=128, BK=64; 8 waves as 2(M) x 4(N); each wave owns a
//   64x32 output tile = 4x2 accumulators; 16 v_wmma per K-stage (2 k-halves).
//   Weights [Co][K]: ONE Tensor-Data-Mover DMA per stage (wave 0 issues,
//   TENSORcnt-tracked, double-buffered, overlapped with WMMA). Activations:
//   gathered with an incremental (ci,kh,kw) walk (1 division per stage) and
//   written to LDS as 4x b128 stores per thread.
// ---------------------------------------------------------------------------
template<int KH, int KW, int STRIDE, int PAD>
__global__ __launch_bounds__(256)
void k_conv_wmma(const __bf16* __restrict__ Aq,   // activations NCHW (bf16)
                 const __bf16* __restrict__ Wq,   // weights [Co][Kdim] (bf16)
                 float* __restrict__ Out,         // output NCHW (f32)
                 int Ci, int IH, int IW, int Co, int OH, int OW, int Kdim)
{
    constexpr int KHW = KH * KW;
    const int tid  = threadIdx.x;
    const int lane = tid & 31;
    const int wave = tid >> 5;
    const int wm   = wave & 1;      // 2 wave rows  -> 64 M-rows each
    const int wn   = wave >> 1;     // 4 wave cols  -> 32 N-cols each
    const int lr   = lane & 15;
    const int hi   = lane >> 4;
    const int m0   = blockIdx.x * 128;
    const int co0  = blockIdx.y * 128;
    const int OHW  = OH * OW;

    __shared__ __align__(16) __bf16 As[2][128 * 64];   // [buf][m][k]   16KB each
    __shared__ __align__(16) __bf16 Bs[2][128 * 64];   // [buf][co][k]  16KB each

    v8f acc[4][2];
    #pragma unroll
    for (int i = 0; i < 4; ++i)
        #pragma unroll
        for (int j = 0; j < 2; ++j)
            acc[i][j] = (v8f)(0.0f);

    // ---- loop-invariant staging assignments ----
    // A: thread -> (row, 32-wide k half); 128 rows x 2 halves
    const int a_row = tid >> 1;
    const int a_k0  = (tid & 1) << 5;
    const int mA     = m0 + a_row;
    const int a_nimg = mA / OHW;
    const int a_rem  = mA - a_nimg * OHW;
    const int a_oh   = a_rem / OW;
    const int a_ow   = a_rem - a_oh * OW;
    const int ih_base = a_oh * STRIDE - PAD;
    const int iw_base = a_ow * STRIDE - PAD;
    const __bf16* Abase = Aq + (size_t)a_nimg * Ci * IH * IW;
    // B: block-uniform weight tile base for the TDM
    const __bf16* WgBase = Wq + (size_t)co0 * Kdim;

    auto stageA = [&](int kk, int p) {
        __align__(16) __bf16 tmp[32];
        if constexpr (KHW == 1) {
            // 1x1 conv: k == ci, fixed spatial point; guard hoisted
            const bool ok = (unsigned)ih_base < (unsigned)IH &&
                            (unsigned)iw_base < (unsigned)IW;
            const int IHW = IH * IW;
            size_t base = (size_t)(kk + a_k0) * IHW +
                          (size_t)ih_base * IW + iw_base;
            #pragma unroll
            for (int j = 0; j < 32; ++j)
                tmp[j] = ok ? Abase[base + (size_t)j * IHW] : (__bf16)0.0f;
        } else {
            // incremental (ci,kh,kw) walk: one division per stage
            int kg0 = kk + a_k0;
            int ci  = kg0 / KHW;
            int r2  = kg0 - ci * KHW;
            int kh  = r2 / KW;
            int kw  = r2 - kh * KW;
            #pragma unroll
            for (int j = 0; j < 32; ++j) {
                int ihp = ih_base + kh;
                int iwp = iw_base + kw;
                __bf16 v = (__bf16)0.0f;
                if ((unsigned)ihp < (unsigned)IH && (unsigned)iwp < (unsigned)IW)
                    v = Abase[((size_t)ci * IH + ihp) * IW + iwp];
                tmp[j] = v;
                if (++kw == KW) { kw = 0; if (++kh == KH) { kh = 0; ++ci; } }
            }
        }
        uint4* dst = (uint4*)&As[p][a_row * 64 + a_k0];
        #pragma unroll
        for (int q = 0; q < 4; ++q)
            dst[q] = *(const uint4*)&tmp[8 * q];
    };

    const int S = Kdim >> 6;        // K is always a multiple of 64 here
    stageA(0, 0);
    if (wave == 0) {
        tdm_load_2d_bf16(&Bs[0][0], WgBase, 64, 128, Kdim);
        wait_tensorcnt0();
    }
    __syncthreads();

    for (int s = 0; s < S; ++s) {
        const int p = s & 1;
        if (s + 1 < S) {                         // overlap with WMMA below
            stageA((s + 1) << 6, p ^ 1);
            if (wave == 0)
                tdm_load_2d_bf16(&Bs[p ^ 1][0], WgBase + ((s + 1) << 6),
                                 64, 128, Kdim);
        }

        #pragma unroll
        for (int kb = 0; kb < 2; ++kb) {
            // fragments per the 16-bit WMMA per-lane layout:
            // lanes 0-15: K 0..7 in v0..3, K 16..23 in v4..7; lanes 16-31: +8
            FragBF afrag[4], bfrag[2];
            #pragma unroll
            for (int mi = 0; mi < 4; ++mi) {
                const __bf16* ap =
                    &As[p][(wm * 64 + mi * 16 + lr) * 64 + kb * 32 + hi * 8];
                afrag[mi].u.lo = *(const uint4*)(ap);
                afrag[mi].u.hi = *(const uint4*)(ap + 16);
            }
            #pragma unroll
            for (int ni = 0; ni < 2; ++ni) {
                const __bf16* bp =
                    &Bs[p][(wn * 32 + ni * 16 + lr) * 64 + kb * 32 + hi * 8];
                bfrag[ni].u.lo = *(const uint4*)(bp);
                bfrag[ni].u.hi = *(const uint4*)(bp + 16);
            }
            #pragma unroll
            for (int mi = 0; mi < 4; ++mi)
                #pragma unroll
                for (int ni = 0; ni < 2; ++ni)
                    acc[mi][ni] = __builtin_amdgcn_wmma_f32_16x16x32_bf16(
                        false, afrag[mi].v, false, bfrag[ni].v,
                        (short)0, acc[mi][ni], false, false);
        }

        if (wave == 0) wait_tensorcnt0();        // next buffer's DMA landed
        __syncthreads();                         // publish to all waves
    }

    // ---- store C (VGPR r: M = r + 8*hi); division hoisted per 8-row group ----
    const int CoOHW = Co * OHW;
    #pragma unroll
    for (int mi = 0; mi < 4; ++mi) {
        const int mbase = m0 + wm * 64 + mi * 16 + hi * 8;
        int nimg = mbase / OHW;
        int rem  = mbase - nimg * OHW;
        size_t base0 = (size_t)nimg * CoOHW;
        #pragma unroll
        for (int r = 0; r < 8; ++r) {
            #pragma unroll
            for (int ni = 0; ni < 2; ++ni) {
                const int colg = co0 + wn * 32 + ni * 16 + lr;
                Out[base0 + (size_t)colg * OHW + rem] = acc[mi][ni][r];
            }
            if (++rem == OHW) { rem = 0; base0 += (size_t)CoOHW; }
        }
    }
}

// ---------------------------------------------------------------------------
// Training-mode BN statistics: one block per channel, reduce over (N,H,W).
// Nested loops -> no per-element division.
// ---------------------------------------------------------------------------
__global__ __launch_bounds__(256)
void k_channel_stats(const float* __restrict__ y, float* __restrict__ mstd,
                     int C, int Nimg, int HW)
{
    const int c = blockIdx.x;
    const int tid = threadIdx.x;
    float s1 = 0.f, s2 = 0.f;
    for (int nimg = 0; nimg < Nimg; ++nimg) {
        const float* yp = y + ((size_t)nimg * C + c) * HW;
        for (int hw = tid; hw < HW; hw += 256) {
            float v = yp[hw];
            s1 += v; s2 += v * v;
        }
    }
    __shared__ float r1[256], r2[256];
    r1[tid] = s1; r2[tid] = s2;
    __syncthreads();
    for (int s = 128; s > 0; s >>= 1) {
        if (tid < s) { r1[tid] += r1[tid + s]; r2[tid] += r2[tid + s]; }
        __syncthreads();
    }
    if (tid == 0) {
        float inv  = 1.0f / (float)(Nimg * HW);
        float mean = r1[0] * inv;
        float var  = r2[0] * inv - mean * mean;
        mstd[2 * c]     = mean;
        mstd[2 * c + 1] = rsqrtf(var + 1e-5f);
    }
}

// ---------------------------------------------------------------------------
// Fused epilogue: out = relu(BN2(conv2) + BNsc(shortcut)). 32-bit indexing.
// ---------------------------------------------------------------------------
__global__ __launch_bounds__(256)
void k_bn_add_relu(const float* __restrict__ ya, const float* __restrict__ yb,
                   const float* __restrict__ msa, const float* __restrict__ ga,
                   const float* __restrict__ ba,
                   const float* __restrict__ msb, const float* __restrict__ gb,
                   const float* __restrict__ bb,
                   float* __restrict__ out, unsigned n, unsigned C, unsigned HW)
{
    unsigned i = blockIdx.x * blockDim.x + threadIdx.x;
    if (i >= n) return;
    unsigned c = (i / HW) % C;
    float a = (ya[i] - msa[2*c]) * msa[2*c+1] * ga[c] + ba[c];
    float b = (yb[i] - msb[2*c]) * msb[2*c+1] * gb[c] + bb[c];
    out[i] = fmaxf(a + b, 0.f);
}

// ---------------------------------------------------------------------------
// Orchestration
// ---------------------------------------------------------------------------
extern "C" void kernel_launch(void* const* d_in, const int* in_sizes, int n_in,
                              void* d_out, int out_size, void* d_ws, size_t ws_size,
                              hipStream_t stream)
{
    const float* x   = (const float*)d_in[0];
    const float* w1  = (const float*)d_in[1];
    const float* g1  = (const float*)d_in[2];
    const float* b1  = (const float*)d_in[3];
    const float* w2  = (const float*)d_in[4];
    const float* g2  = (const float*)d_in[5];
    const float* b2  = (const float*)d_in[6];
    const float* wsc = (const float*)d_in[7];
    const float* gsc = (const float*)d_in[8];
    const float* bsc = (const float*)d_in[9];
    float* out = (float*)d_out;
    (void)in_sizes; (void)n_in; (void)out_size; (void)ws_size;

    const int Nimg = 64, Ci = 128, H = 56, Co = 256, OH = 28;
    const unsigned xN   = (unsigned)(Nimg * Ci * H * H);     // 25,690,112
    const unsigned w1N  = (unsigned)(Co * Ci * 9);           //    294,912
    const unsigned w2N  = (unsigned)(Co * Co * 9);           //    589,824
    const unsigned wscN = (unsigned)(Co * Ci);               //     32,768
    const unsigned yN   = (unsigned)(Nimg * Co * OH * OH);   // 12,845,056

    char* ws = (char*)d_ws;
    size_t off = 0;
    auto carve = [&](size_t bytes) -> void* {
        void* p = ws + off;
        off += (bytes + 255) & ~(size_t)255;
        return p;
    };
    __bf16* xq    = (__bf16*)carve((size_t)xN  * 2);
    __bf16* w1q   = (__bf16*)carve((size_t)w1N * 2);
    __bf16* w2q   = (__bf16*)carve((size_t)w2N * 2);
    __bf16* wscq  = (__bf16*)carve((size_t)wscN * 2);
    __bf16* a1q   = (__bf16*)carve((size_t)yN  * 2);
    float*  c1raw = (float*)carve((size_t)yN * 4);
    float*  c2raw = (float*)carve((size_t)yN * 4);
    float*  scraw = (float*)carve((size_t)yN * 4);
    float*  st1   = (float*)carve(2 * Co * 4);
    float*  st2   = (float*)carve(2 * Co * 4);
    float*  stsc  = (float*)carve(2 * Co * 4);

    const dim3 blk(256);
    auto qgrid = [](unsigned n) {
        return dim3((((n + 63) / 64) + 7) / 8);   // 8 waves/TB, 1 block/wave
    };

    // 1) HBFP8-quantize activations and weights into bf16 (exact).
    k_bfp_quant<<<qgrid(xN),   blk, 0, stream>>>(x,   xq,   xN,   nullptr, nullptr, nullptr, 1, 1, 0);
    k_bfp_quant<<<qgrid(w1N),  blk, 0, stream>>>(w1,  w1q,  w1N,  nullptr, nullptr, nullptr, 1, 1, 0);
    k_bfp_quant<<<qgrid(w2N),  blk, 0, stream>>>(w2,  w2q,  w2N,  nullptr, nullptr, nullptr, 1, 1, 0);
    k_bfp_quant<<<qgrid(wscN), blk, 0, stream>>>(wsc, wscq, wscN, nullptr, nullptr, nullptr, 1, 1, 0);

    // 2) conv1 (3x3, s2, p1): M=50176, N=256, K=1152
    const dim3 cgrid(392, 2);   // M/128, Co/128
    k_conv_wmma<3,3,2,1><<<cgrid, blk, 0, stream>>>(xq, w1q, c1raw,
        Ci, H, H, Co, OH, OH, Ci * 9);

    // 3) BN1 stats, then fused BN1+ReLU+HBFP8-requantize -> conv2 input
    k_channel_stats<<<dim3(Co), blk, 0, stream>>>(c1raw, st1, Co, Nimg, OH * OH);
    k_bfp_quant<<<qgrid(yN), blk, 0, stream>>>(c1raw, a1q, yN, st1, g1, b1,
                                               (unsigned)Co, (unsigned)(OH * OH), 1);

    // 4) conv2 (3x3, s1, p1): K=2304
    k_conv_wmma<3,3,1,1><<<cgrid, blk, 0, stream>>>(a1q, w2q, c2raw,
        Co, OH, OH, Co, OH, OH, Co * 9);

    // 5) shortcut conv (1x1, s2, p0): K=128
    k_conv_wmma<1,1,2,0><<<cgrid, blk, 0, stream>>>(xq, wscq, scraw,
        Ci, H, H, Co, OH, OH, Ci);

    // 6) BN stats for both branches, then fused add + ReLU
    k_channel_stats<<<dim3(Co), blk, 0, stream>>>(c2raw, st2,  Co, Nimg, OH * OH);
    k_channel_stats<<<dim3(Co), blk, 0, stream>>>(scraw, stsc, Co, Nimg, OH * OH);
    k_bn_add_relu<<<dim3((yN + 255) / 256), blk, 0, stream>>>(
        c2raw, scraw, st2, g2, b2, stsc, gsc, bsc, out, yN,
        (unsigned)Co, (unsigned)(OH * OH));
}
